// Computation_Block_2568390443192
// MI455X (gfx1250) — compile-verified, matrix operand order checked by reading, not run on hardware
//
#include <hip/hip_runtime.h>
#include <math.h>

typedef __bf16 bf16_t;
typedef __attribute__((ext_vector_type(8)))  bf16_t v8bf;
typedef __attribute__((ext_vector_type(16))) bf16_t v16bf;
typedef __attribute__((ext_vector_type(8)))  float  v8f;

#define LCOUNT 2
#define DMODEL 512
#define HDIM   2048
#define QKD    128
#define GSZN   256
#define KSZ    17
#define BB     2
#define SS     8192
#define MTOT   (BB*SS)          // 16384 rows

// ---------------- WMMA GEMM: C[M,N] = epi( A[M,K] * B^T[N,K] ) ----------------
#define BM 64
#define BN 128
#define BKK 32
#define LDSP 40   // padded LDS row stride (elems): conflict-free, 16B aligned

__device__ __forceinline__ v16bf frag_ld(const bf16_t* base, int row, int hi) {
  union { v16bf v; v8bf h[2]; } f;
  const bf16_t* p = base + row * LDSP;
  f.h[0] = *(const v8bf*)(p + hi * 8);        // K = hi*8 .. hi*8+7
  f.h[1] = *(const v8bf*)(p + 16 + hi * 8);   // K = 16+hi*8 .. 16+hi*8+7
  return f.v;
}

// async 16B global -> LDS copy (CDNA5, tracked by ASYNCcnt)
__device__ __forceinline__ void async_b128(bf16_t* lds_dst, const bf16_t* gsrc) {
  unsigned loff = (unsigned)(unsigned long long)lds_dst;   // low 32 bits = LDS offset
  asm volatile("global_load_async_to_lds_b128 %0, %1, off"
               :: "v"(loff), "v"(gsrc) : "memory");
}

// EPI: 0 = store, 1 = silu(acc+bias[n]), 2 = relu(acc*scale)^2, 3 = acc*scale, 4 = C += acc
template<int EPI, typename TO>
__global__ __launch_bounds__(256) void k_gemm(
    const bf16_t* __restrict__ A, const bf16_t* __restrict__ B, TO* __restrict__ C,
    int K, int lda, int ldb, int ldc,
    long sA, long sBo, long sBi, long sC, int gpb,
    const float* __restrict__ bias, float scale)
{
  __shared__ bf16_t As[2][BM * LDSP];
  __shared__ bf16_t Bs[2][BN * LDSP];
  const int z = blockIdx.z;
  const int m0 = blockIdx.y * BM;
  const int n0 = blockIdx.x * BN;
  const int tid  = threadIdx.x;
  const int lane = tid & 31, wave = tid >> 5;
  const int wm = wave >> 2, wn = wave & 3;
  const int hi = lane >> 4, lr = lane & 15;

  // per-thread 16B-chunk sources (A: 256 chunks = 1/thread; B: 512 chunks = 2/thread)
  const int ar  = tid >> 2, ac = (tid & 3) * 8;
  const int br0 = tid >> 2, br1 = (tid + 256) >> 2;
  const bf16_t* Ag  = A + (long)z * sA + (long)(m0 + ar) * lda + ac;
  const bf16_t* Bb  = B + (long)(z / gpb) * sBo + (long)(z % gpb) * sBi;
  const bf16_t* Bg0 = Bb + (long)(n0 + br0) * ldb + ac;
  const bf16_t* Bg1 = Bb + (long)(n0 + br1) * ldb + ac;
  bf16_t* Asl  = &As[0][0] + ar  * LDSP + ac;
  bf16_t* Bsl0 = &Bs[0][0] + br0 * LDSP + ac;
  bf16_t* Bsl1 = &Bs[0][0] + br1 * LDSP + ac;
  TO* Cb = C + (long)z * sC;

  const int NT = K / BKK;
  // prologue: async-stage tile 0 into buffer 0
  async_b128(Asl,  Ag);
  async_b128(Bsl0, Bg0);
  async_b128(Bsl1, Bg1);

  v8f acc[2][2] = {};
  for (int kt = 0; kt < NT; ++kt) {
    const int cur = kt & 1;
    if (kt + 1 < NT) {          // issue next tile into other buffer, then allow 3 outstanding
      const int nxt = cur ^ 1;
      long ko = (long)(kt + 1) * BKK;
      async_b128(Asl  + nxt * (BM * LDSP), Ag  + ko);
      async_b128(Bsl0 + nxt * (BN * LDSP), Bg0 + ko);
      async_b128(Bsl1 + nxt * (BN * LDSP), Bg1 + ko);
      asm volatile("s_wait_asynccnt 0x3" ::: "memory");
    } else {
      asm volatile("s_wait_asynccnt 0x0" ::: "memory");
    }
    __syncthreads();            // tile `cur` visible to all waves

    const bf16_t* Ap = &As[cur][0];
    const bf16_t* Bp = &Bs[cur][0];
    v16bf af[2], bfm[2];
    #pragma unroll
    for (int i = 0; i < 2; i++) af[i]  = frag_ld(Ap, wm * 32 + i * 16 + lr, hi);
    #pragma unroll
    for (int j = 0; j < 2; j++) bfm[j] = frag_ld(Bp, wn * 32 + j * 16 + lr, hi);
    #pragma unroll
    for (int i = 0; i < 2; i++)
      #pragma unroll
      for (int j = 0; j < 2; j++)
        acc[i][j] = __builtin_amdgcn_wmma_f32_16x16x32_bf16(
            false, af[i], false, bfm[j], (short)0, acc[i][j], false, false);
    __syncthreads();            // done reading `cur` before it is overwritten
  }

  #pragma unroll
  for (int i = 0; i < 2; i++)
    #pragma unroll
    for (int j = 0; j < 2; j++) {
      int n  = n0 + wn * 32 + j * 16 + lr;
      int mb = m0 + wm * 32 + i * 16 + hi * 8;
      float bv = 0.f;
      if constexpr (EPI == 1) bv = bias[n];
      #pragma unroll
      for (int r = 0; r < 8; r++) {
        float v = acc[i][j][r];
        long oi = (long)(mb + r) * ldc + n;
        if constexpr (EPI == 1) { float t = v + bv; v = t / (1.f + __expf(-t)); }
        else if constexpr (EPI == 2) { float t = v * scale; t = t > 0.f ? t : 0.f; v = t * t; }
        else if constexpr (EPI == 3) { v *= scale; }
        else if constexpr (EPI == 4) { v += (float)Cb[oi]; }
        Cb[oi] = (TO)v;
      }
    }
}

// ---------------- tiled transpose (+optional bf16 convert) ----------------
// out[c*ldout + r] = in[r*ldin + colOff + c]
template<typename TO>
__global__ void k_tc(const float* __restrict__ in, TO* __restrict__ out,
                     long inB, long outB, int ldin, int ldout, int colOff)
{
  __shared__ float t[32][33];
  int b = blockIdx.z;
  int r0 = blockIdx.x * 32, c0 = blockIdx.y * 32;
  const float* src = in + (long)b * inB;
  TO* dst = out + (long)b * outB;
  #pragma unroll
  for (int k = 0; k < 4; k++) {
    int r = r0 + threadIdx.y + k * 8;
    t[threadIdx.y + k * 8][threadIdx.x] = src[(long)r * ldin + colOff + c0 + threadIdx.x];
  }
  __syncthreads();
  #pragma unroll
  for (int k = 0; k < 4; k++) {
    int c = c0 + threadIdx.y + k * 8;
    dst[(long)c * ldout + r0 + threadIdx.x] = (TO)t[threadIdx.x][threadIdx.y + k * 8];
  }
}

// ---------------- elementwise / reduction kernels ----------------
__global__ void k_shift(const float* __restrict__ h, float* __restrict__ nx) {
  long i = (long)blockIdx.x * 256 + threadIdx.x;   // over BB*SS*DMODEL
  long c = i % DMODEL;
  long s = (i / DMODEL) % SS;
  float v;
  if (c < DMODEL / 2) v = (s == 0) ? 0.f : h[i - DMODEL];
  else                v = h[i];
  nx[i] = v;
}

__global__ void k_rownorm(const float* __restrict__ in, bf16_t* __restrict__ out,
                          int C, const float* __restrict__ gptr, int gidx) {
  long row = blockIdx.x;
  const float* x = in + row * (long)C;
  __shared__ float red[256];
  float ss = 0.f;
  for (int j = threadIdx.x; j < C; j += 256) { float v = x[j]; ss += v * v; }
  red[threadIdx.x] = ss; __syncthreads();
  for (int s = 128; s > 0; s >>= 1) {
    if (threadIdx.x < s) red[threadIdx.x] += red[threadIdx.x + s];
    __syncthreads();
  }
  float rms = sqrtf(red[0] / (float)C);
  float inv = gptr[gidx] / fmaxf(rms, 1e-5f);
  for (int j = threadIdx.x; j < C; j += 256) out[row * (long)C + j] = (bf16_t)(x[j] * inv);
}

// out = in + depthwise_conv(in, k); accumulate ? out += ... : out = ...
__global__ void k_dwconv(const float* __restrict__ in, float* __restrict__ out,
                         const float* __restrict__ kern, int C, int accumulate) {
  int c = blockIdx.x * blockDim.x + threadIdx.x;
  int s = blockIdx.y, b = blockIdx.z;
  float kk[KSZ];
  #pragma unroll
  for (int j = 0; j < KSZ; j++) kk[j] = kern[(long)c * KSZ + j];
  const float* base = in + (long)b * SS * C;
  float acc = base[(long)s * C + c];
  #pragma unroll
  for (int j = 0; j < KSZ; j++) {
    int ss = s + j - (KSZ - 1) / 2;
    if (0 <= ss && ss < SS) acc += kk[j] * base[(long)ss * C + c];
  }
  long oi = ((long)b * SS + s) * (long)C + c;
  if (accumulate) out[oi] += acc; else out[oi] = acc;
}

__global__ void k_heads(const float* __restrict__ qkc,
                        const float* __restrict__ gamma, const float* __restrict__ beta,
                        bf16_t* __restrict__ quadq, bf16_t* __restrict__ linq,
                        bf16_t* __restrict__ qkkbf, float* __restrict__ linkk) {
  int d = threadIdx.x;                       // 0..127
  int s = blockIdx.x, b = blockIdx.y;
  long row = (long)b * SS + s;
  float xd = qkc[row * QKD + d];
  int dp = d ^ 1;
  float xp = qkc[row * QKD + dp];
  float cs = 1.f, sn = 0.f;
  if (d < 32) {
    int iF = d >> 1;
    float inv = powf(10000.f, -(float)(2 * iF) / 32.f);
    float ang = (float)s * inv;
    cs = cosf(ang); sn = sinf(ang);
  }
  #pragma unroll
  for (int t = 0; t < 4; t++) {
    float yd = xd * gamma[t * QKD + d] + beta[t * QKD + d];
    float o;
    if (d < 32) {
      float yp = xp * gamma[t * QKD + dp] + beta[t * QKD + dp];
      o = (d & 1) ? (yd * cs + yp * sn) : (yd * cs - yp * sn);
    } else o = yd;
    if (t == 0) quadq[row * QKD + d] = (bf16_t)o;
    else if (t == 1) linq[row * QKD + d] = (bf16_t)o;
    else if (t == 2) qkkbf[row * QKD + d] = (bf16_t)o;
    else linkk[row * QKD + d] = o;
  }
}

__global__ void k_gate(const float* __restrict__ hidc, const float* __restrict__ qv,
                       const float* __restrict__ qu, float* __restrict__ outg) {
  long i = (long)blockIdx.x * 256 + threadIdx.x;   // over BB*SS*1024
  long row = i >> 10, e = i & 1023;
  float v = hidc[row * HDIM + e];
  float u = hidc[row * HDIM + 1024 + e];
  float av = qv[i], au = qu[i];
  outg[i] = au * v * (1.f / (1.f + __expf(-av * u)));
}

__global__ void k_layernorm(const float* __restrict__ in, float* __restrict__ out,
                            const float* __restrict__ g, const float* __restrict__ b) {
  long row = blockIdx.x;
  const float* x = in + row * (long)DMODEL;
  __shared__ float rs[256], rs2[256];
  float s = 0.f, s2 = 0.f;
  for (int j = threadIdx.x; j < DMODEL; j += 256) { float v = x[j]; s += v; s2 += v * v; }
  rs[threadIdx.x] = s; rs2[threadIdx.x] = s2; __syncthreads();
  for (int k = 128; k > 0; k >>= 1) {
    if (threadIdx.x < k) { rs[threadIdx.x] += rs[threadIdx.x + k]; rs2[threadIdx.x] += rs2[threadIdx.x + k]; }
    __syncthreads();
  }
  float mean = rs[0] / (float)DMODEL;
  float var  = rs2[0] / (float)DMODEL - mean * mean;
  float r = rsqrtf(var + 1e-6f);
  for (int j = threadIdx.x; j < DMODEL; j += 256)
    out[row * (long)DMODEL + j] = (x[j] - mean) * r * g[j] + b[j];
}

__global__ void k_stat1(const float* __restrict__ y, float* __restrict__ part, long chunk) {
  int b = blockIdx.y;
  long base = (long)b * SS * DMODEL + (long)blockIdx.x * chunk;
  float s = 0.f, s2 = 0.f;
  for (long i = threadIdx.x; i < chunk; i += 256) { float v = y[base + i]; s += v; s2 += v * v; }
  __shared__ float rs[256], rs2[256];
  rs[threadIdx.x] = s; rs2[threadIdx.x] = s2; __syncthreads();
  for (int k = 128; k > 0; k >>= 1) {
    if (threadIdx.x < k) { rs[threadIdx.x] += rs[threadIdx.x + k]; rs2[threadIdx.x] += rs2[threadIdx.x + k]; }
    __syncthreads();
  }
  if (threadIdx.x == 0) {
    part[((long)b * gridDim.x + blockIdx.x) * 2]     = rs[0];
    part[((long)b * gridDim.x + blockIdx.x) * 2 + 1] = rs2[0];
  }
}

__global__ void k_stat2(const float* __restrict__ part, float* __restrict__ stats, int nblk) {
  int b = blockIdx.x;
  __shared__ float rs[256], rs2[256];
  float s = 0.f, s2 = 0.f;
  for (int i = threadIdx.x; i < nblk; i += 256) {
    s  += part[((long)b * nblk + i) * 2];
    s2 += part[((long)b * nblk + i) * 2 + 1];
  }
  rs[threadIdx.x] = s; rs2[threadIdx.x] = s2; __syncthreads();
  for (int k = 128; k > 0; k >>= 1) {
    if (threadIdx.x < k) { rs[threadIdx.x] += rs[threadIdx.x + k]; rs2[threadIdx.x] += rs2[threadIdx.x + k]; }
    __syncthreads();
  }
  if (threadIdx.x == 0) {
    float N = (float)SS * (float)DMODEL;
    float mean = rs[0] / N;
    float var  = rs2[0] / N - mean * mean;
    stats[b * 2]     = mean;
    stats[b * 2 + 1] = rsqrtf(var + 1e-8f);
  }
}

// out[b,d,s] = (y[b,s,d]-mean)*rstd*gw[d] + gb[d] + x[b,d,s]
__global__ void k_final(const float* __restrict__ y, const float* __restrict__ x,
                        float* __restrict__ out, const float* __restrict__ stats,
                        const float* __restrict__ gw, const float* __restrict__ gb) {
  __shared__ float t[32][33];
  int b = blockIdx.z;
  int d0 = blockIdx.x * 32, s0 = blockIdx.y * 32;
  float mean = stats[b * 2], rstd = stats[b * 2 + 1];
  #pragma unroll
  for (int k = 0; k < 4; k++) {
    int s = s0 + threadIdx.y + k * 8;
    t[threadIdx.y + k * 8][threadIdx.x] = y[((long)b * SS + s) * DMODEL + d0 + threadIdx.x];
  }
  __syncthreads();
  #pragma unroll
  for (int k = 0; k < 4; k++) {
    int d = d0 + threadIdx.y + k * 8;
    int s = s0 + threadIdx.x;
    long xi = ((long)b * DMODEL + d) * (long)SS + s;
    float v = t[threadIdx.x][threadIdx.y + k * 8];
    out[xi] = (v - mean) * rstd * gw[d] + gb[d] + x[xi];
  }
}

// ---------------- host launcher ----------------
extern "C" void kernel_launch(void* const* d_in, const int* in_sizes, int n_in,
                              void* d_out, int out_size, void* d_ws, size_t ws_size,
                              hipStream_t stream) {
  (void)in_sizes; (void)n_in; (void)out_size; (void)ws_size;
  const float* x        = (const float*)d_in[0];
  const float* g_hidden = (const float*)d_in[1];
  const float* W_hidden = (const float*)d_in[2];
  const float* b_hidden = (const float*)d_in[3];
  const float* k_hidden = (const float*)d_in[4];
  const float* g_qk     = (const float*)d_in[5];
  const float* W_qk     = (const float*)d_in[6];
  const float* b_qk     = (const float*)d_in[7];
  const float* k_qk     = (const float*)d_in[8];
  const float* g_out    = (const float*)d_in[9];
  const float* W_out    = (const float*)d_in[10];
  const float* b_out    = (const float*)d_in[11];
  const float* k_out    = (const float*)d_in[12];
  const float* gamma_qk = (const float*)d_in[13];
  const float* beta_qk  = (const float*)d_in[14];
  const float* ln_g     = (const float*)d_in[15];
  const float* ln_b     = (const float*)d_in[16];
  const float* gn_w     = (const float*)d_in[17];
  const float* gn_b     = (const float*)d_in[18];
  float* out = (float*)d_out;

  char* p = (char*)d_ws;
  auto alloc = [&](size_t bytes) -> char* {
    char* r = p; p += (bytes + 255) & ~(size_t)255; return r;
  };
  float*  h      = (float*)alloc((size_t)MTOT * DMODEL * 4);
  float*  nx     = (float*)alloc((size_t)MTOT * DMODEL * 4);   // reused as y-buf at end
  bf16_t* normb  = (bf16_t*)alloc((size_t)MTOT * 1024 * 2);
  float*  hid    = (float*)alloc((size_t)MTOT * HDIM * 4);     // reused as qv/qu
  float*  hidc   = (float*)alloc((size_t)MTOT * HDIM * 4);
  float*  qk     = (float*)alloc((size_t)MTOT * QKD * 4);
  float*  qkc    = (float*)alloc((size_t)MTOT * QKD * 4);
  bf16_t* quadq  = (bf16_t*)alloc((size_t)MTOT * QKD * 2);
  bf16_t* linq   = (bf16_t*)alloc((size_t)MTOT * QKD * 2);
  float*  linkk  = (float*)alloc((size_t)MTOT * QKD * 4);
  bf16_t* attn   = (bf16_t*)alloc((size_t)BB * 32 * GSZN * GSZN * 2);
  float*  outg   = (float*)alloc((size_t)MTOT * 1024 * 4);
  float*  proj   = (float*)alloc((size_t)MTOT * DMODEL * 4);
  float*  part   = (float*)alloc((size_t)BB * 256 * 2 * 4);
  float*  stats  = (float*)alloc(16);
  bf16_t* wth    = (bf16_t*)alloc((size_t)HDIM * DMODEL * 2);
  bf16_t* wtq    = (bf16_t*)alloc((size_t)QKD * DMODEL * 2);
  bf16_t* wto    = (bf16_t*)alloc((size_t)DMODEL * 1024 * 2);
  bf16_t* qkkbf  = (bf16_t*)alloc((size_t)MTOT * QKD * 2);
  bf16_t* vvT    = (bf16_t*)alloc((size_t)BB * 1024 * SS * 2);
  bf16_t* uuT    = (bf16_t*)alloc((size_t)BB * 1024 * SS * 2);
  bf16_t* lkT    = (bf16_t*)alloc((size_t)BB * QKD * SS * 2);
  bf16_t* lkvb   = (bf16_t*)alloc((size_t)BB * 1024 * QKD * 2);
  bf16_t* lkub   = (bf16_t*)alloc((size_t)BB * 1024 * QKD * 2);
  float*  qv = hid;
  float*  qu = hid + (size_t)MTOT * 1024;

  dim3 tb(32, 8);
  // x (B,D,S) -> h (B,S,D)
  k_tc<float><<<dim3(DMODEL/32, SS/32, BB), tb, 0, stream>>>(
      x, h, (long)DMODEL*SS, (long)SS*DMODEL, SS, DMODEL, 0);

  for (int i = 0; i < LCOUNT; i++) {
    // weight transposes -> bf16 [N][K]
    k_tc<bf16_t><<<dim3(DMODEL/32, HDIM/32, 1), tb, 0, stream>>>(
        W_hidden + (size_t)i*DMODEL*HDIM, wth, 0, 0, HDIM, DMODEL, 0);
    k_tc<bf16_t><<<dim3(DMODEL/32, QKD/32, 1), tb, 0, stream>>>(
        W_qk + (size_t)i*DMODEL*QKD, wtq, 0, 0, QKD, DMODEL, 0);
    k_tc<bf16_t><<<dim3(1024/32, DMODEL/32, 1), tb, 0, stream>>>(
        W_out + (size_t)i*1024*DMODEL, wto, 0, 0, DMODEL, 1024, 0);

    // shift-concat, scalenorm (bf16 out)
    k_shift<<<(MTOT*DMODEL)/256, 256, 0, stream>>>(h, nx);
    k_rownorm<<<MTOT, 256, 0, stream>>>(nx, normb, DMODEL, g_hidden, i);

    // hidden GEMM + silu, then dwconv
    k_gemm<1,float><<<dim3(HDIM/BN, MTOT/BM, 1), 256, 0, stream>>>(
        normb, wth, hid, DMODEL, DMODEL, DMODEL, HDIM, 0,0,0,0, 1,
        b_hidden + (size_t)i*HDIM, 0.f);
    k_dwconv<<<dim3(HDIM/256, SS, BB), 256, 0, stream>>>(
        hid, hidc, k_hidden + (size_t)i*HDIM*KSZ, HDIM, 0);

    // qk GEMM + silu, dwconv, heads+rotary
    k_gemm<1,float><<<dim3(QKD/BN, MTOT/BM, 1), 256, 0, stream>>>(
        normb, wtq, qk, DMODEL, DMODEL, DMODEL, QKD, 0,0,0,0, 1,
        b_qk + (size_t)i*QKD, 0.f);
    k_dwconv<<<dim3(1, SS, BB), 128, 0, stream>>>(
        qk, qkc, k_qk + (size_t)i*QKD*KSZ, QKD, 0);
    k_heads<<<dim3(SS, BB), QKD, 0, stream>>>(
        qkc, gamma_qk + (size_t)i*4*QKD, beta_qk + (size_t)i*4*QKD,
        quadq, linq, qkkbf, linkk);

    // transposed bf16 operands: v^T, u^T, lin_k^T
    k_tc<bf16_t><<<dim3(SS/32, 1024/32, BB), tb, 0, stream>>>(
        hidc, vvT, (long)SS*HDIM, (long)1024*SS, HDIM, SS, 0);
    k_tc<bf16_t><<<dim3(SS/32, 1024/32, BB), tb, 0, stream>>>(
        hidc, uuT, (long)SS*HDIM, (long)1024*SS, HDIM, SS, 1024);
    k_tc<bf16_t><<<dim3(SS/32, QKD/32, BB), tb, 0, stream>>>(
        linkk, lkT, (long)SS*QKD, (long)QKD*SS, QKD, SS, 0);

    // quadratic attention: scores (relu^2 -> bf16), attn@v, attn@u
    k_gemm<2,bf16_t><<<dim3(GSZN/BN, GSZN/BM, BB*32), 256, 0, stream>>>(
        quadq, qkkbf, attn, QKD, QKD, QKD, GSZN,
        (long)GSZN*QKD, (long)GSZN*QKD, 0, (long)GSZN*GSZN, 1, nullptr, 1.f/GSZN);
    k_gemm<0,float><<<dim3(1024/BN, GSZN/BM, BB*32), 256, 0, stream>>>(
        attn, vvT, qv, GSZN, GSZN, SS, 1024,
        (long)GSZN*GSZN, (long)1024*SS, GSZN, (long)GSZN*1024, 32, nullptr, 0.f);
    k_gemm<0,float><<<dim3(1024/BN, GSZN/BM, BB*32), 256, 0, stream>>>(
        attn, uuT, qu, GSZN, GSZN, SS, 1024,
        (long)GSZN*GSZN, (long)1024*SS, GSZN, (long)GSZN*1024, 32, nullptr, 0.f);

    // linear attention: lkv^T = v^T @ lin_k / S (bf16 out); then lq @ lkv += qv/qu
    k_gemm<3,bf16_t><<<dim3(QKD/BN, 1024/BM, BB), 256, 0, stream>>>(
        vvT, lkT, lkvb, SS, SS, SS, QKD,
        (long)1024*SS, (long)QKD*SS, 0, (long)1024*QKD, 1, nullptr, 1.f/(float)SS);
    k_gemm<3,bf16_t><<<dim3(QKD/BN, 1024/BM, BB), 256, 0, stream>>>(
        uuT, lkT, lkub, SS, SS, SS, QKD,
        (long)1024*SS, (long)QKD*SS, 0, (long)1024*QKD, 1, nullptr, 1.f/(float)SS);
    k_gemm<4,float><<<dim3(1024/BN, SS/BM, BB), 256, 0, stream>>>(
        linq, lkvb, qv, QKD, QKD, QKD, 1024,
        (long)SS*QKD, (long)1024*QKD, 0, (long)SS*1024, 1, nullptr, 0.f);
    k_gemm<4,float><<<dim3(1024/BN, SS/BM, BB), 256, 0, stream>>>(
        linq, lkub, qu, QKD, QKD, QKD, 1024,
        (long)SS*QKD, (long)1024*QKD, 0, (long)SS*1024, 1, nullptr, 0.f);

    // gating, output projection (norm + GEMM + silu + dwconv), residual into h
    k_gate<<<(MTOT*1024)/256, 256, 0, stream>>>(hidc, qv, qu, outg);
    k_rownorm<<<MTOT, 256, 0, stream>>>(outg, normb, 1024, g_out, i);
    k_gemm<1,float><<<dim3(DMODEL/BN, MTOT/BM, 1), 256, 0, stream>>>(
        normb, wto, proj, 1024, 1024, 1024, DMODEL, 0,0,0,0, 1,
        b_out + (size_t)i*DMODEL, 0.f);
    k_dwconv<<<dim3(DMODEL/256, SS, BB), 256, 0, stream>>>(
        proj, h, k_out + (size_t)i*DMODEL*KSZ, DMODEL, 1);
  }

  // final layernorm over D, per-batch groupnorm, transpose, residual with input x
  float* ybuf = nx;
  k_layernorm<<<MTOT, 256, 0, stream>>>(h, ybuf, ln_g, ln_b);
  k_stat1<<<dim3(256, BB), 256, 0, stream>>>(ybuf, part, (long)SS*DMODEL/256);
  k_stat2<<<BB, 256, 0, stream>>>(part, stats, 256);
  k_final<<<dim3(DMODEL/32, SS/32, BB), tb, 0, stream>>>(ybuf, x, out, stats, gn_w, gn_b);
}
